// Attention_21569325760808
// MI455X (gfx1250) — matrix operand order repaired and checked by reading, not compiled, not run on hardware
//
#include <hip/hip_runtime.h>

// ---------------------------------------------------------------------------
// MI455X (gfx1250) attention pipeline: f16 WMMA, f32 accumulate, async-LDS
// double-buffered staging (manually software-pipelined, static buffer ids).
// B=2, S=2048, NX=1024, H=16, D=64.  wave32, 256-thread blocks (8 waves).
// ---------------------------------------------------------------------------

typedef _Float16 half16 __attribute__((ext_vector_type(16)));
typedef _Float16 halfx4 __attribute__((ext_vector_type(4)));
typedef float    floatx8 __attribute__((ext_vector_type(8)));
typedef float    floatx4 __attribute__((ext_vector_type(4)));

union FragH {
  half16  h;
  floatx4 f[2];
};

// CDNA5 async copy: global -> LDS, 16B per lane, tracked by ASYNCcnt.
// INST_OFFSET is added to BOTH the LDS and global addresses (ISA 15.18.3
// op 98), so consecutive chunks share one address pair + immediate offsets.
__device__ __forceinline__ void async_b128(void* lds, const void* g) {
  unsigned l = (unsigned)(size_t)lds;
  asm volatile("global_load_async_to_lds_b128 %0, %1, off"
               :: "v"(l), "v"(g) : "memory");
}
template <int OFF>
__device__ __forceinline__ void async_b128_off(void* lds, const void* g) {
  unsigned l = (unsigned)(size_t)lds;
  asm volatile("global_load_async_to_lds_b128 %0, %1, off offset:%2"
               :: "v"(l), "v"(g), "i"(OFF) : "memory");
}
__device__ __forceinline__ void wait_async0() {
  asm volatile("s_wait_asynccnt 0x0" ::: "memory");
}

// A-operand 16x32 (MxK) from row-major [rows][ld] f16:
// lane l: row = l&15; K = (l>>4)*8 + {0..7}  and  16 + (l>>4)*8 + {0..7}
__device__ __forceinline__ half16 frag_a_rowmajor(const _Float16* p0, int ld,
                                                  int l16, int lh) {
  FragH f;
  const _Float16* p = p0 + l16 * ld + lh * 8;
  f.f[0] = *(const floatx4*)(p);
  f.f[1] = *(const floatx4*)(p + 16);
  return f.h;
}

// B-operand 32x16 (KxN) stored as [N][K] f16:
// lane l: col n = l&15; K = (l>>4)*8 + {0..7}  and  16 + (l>>4)*8 + {0..7}
__device__ __forceinline__ half16 frag_b_nk(const _Float16* p0, int ld,
                                            int l16, int lh) {
  FragH f;
  const _Float16* p = p0 + l16 * ld + lh * 8;
  f.f[0] = *(const floatx4*)(p);
  f.f[1] = *(const floatx4*)(p + 16);
  return f.h;
}

// ---------------------------------------------------------------------------
// LDS-tiled weight transpose + f32->f16:  Wt[n][k] = (f16)W[k][n], K = 1024.
// Grid: (K/32, N/32); both global accesses coalesced.
// ---------------------------------------------------------------------------
__global__ __launch_bounds__(256) void k_transpose_cvt(
    const float* __restrict__ W, _Float16* __restrict__ Wt, int N) {
  __shared__ _Float16 t[32][33];
  const int bk = blockIdx.x;
  const int bn = blockIdx.y;
  const int c = threadIdx.x & 31, r0 = threadIdx.x >> 5;
#pragma unroll
  for (int p = 0; p < 4; ++p) {
    int r = r0 + p * 8;
    t[r][c] = (_Float16)W[(size_t)(bk * 32 + r) * N + bn * 32 + c];
  }
  __syncthreads();
#pragma unroll
  for (int p = 0; p < 4; ++p) {
    int r = r0 + p * 8;
    Wt[(size_t)(bn * 32 + r) * 1024 + bk * 32 + c] = t[c][r];
  }
}

// x (f32) -> f16, 4 elements / thread
__global__ __launch_bounds__(256) void k_cvt_f16(
    const float* __restrict__ src, _Float16* __restrict__ dst, int total4) {
  int i = blockIdx.x * 256 + threadIdx.x;
  if (i >= total4) return;
  floatx4 v = *(const floatx4*)(src + (size_t)i * 4);
  halfx4 h = {(_Float16)v.x, (_Float16)v.y, (_Float16)v.z, (_Float16)v.w};
  *(halfx4*)(dst + (size_t)i * 4) = h;
}

// ---------------------------------------------------------------------------
// QKV GEMM: [4096 x 1024] @ [1024 x 3072] + bias.
// Block tile M=32, N=128; 8 waves each own 16x32 of C (2 accumulators);
// K-step 64; double-buffered async staging, manual 2x pipeline unroll.
// Writes Qh/Kh as [B][H][S][D] f16, V transposed as Vt [B][H][D][S] f16.
// ---------------------------------------------------------------------------
__global__ __launch_bounds__(256) void k_qkv_gemm(
    const _Float16* __restrict__ xh, const _Float16* __restrict__ WqkvT,
    const float* __restrict__ bqkv, _Float16* __restrict__ Qh,
    _Float16* __restrict__ Kh, _Float16* __restrict__ Vt) {
  __shared__ _Float16 As[2][32][64];
  __shared__ _Float16 Bs[2][128][64];
  const int bm = blockIdx.x;          // 128 tiles of M
  const int bn = blockIdx.y;          // 24 tiles of N
  const int tid = threadIdx.x;
  const int lane = tid & 31, wv = tid >> 5;
  const int l16 = lane & 15, lh = lane >> 4;
  const int mT = wv & 1, nT = wv >> 1;          // 2 x 4 waves, 16 x 32 each

  floatx8 acc[2] = {};

  auto stage = [&](int buf, int k0) {
    {  // A: 32x64 halves, one b128 per thread
      int r = tid >> 3, c = (tid & 7) * 8;
      async_b128(&As[buf][r][c],
                 &xh[((size_t)(bm * 32 + r)) * 1024 + k0 + c]);
    }
    {  // B ([n][k]): 128x64 halves, four b128 per thread (shared addr pair)
      int r = tid >> 1, c0 = (tid & 1) * 32;
      const _Float16* g = &WqkvT[((size_t)(bn * 128 + r)) * 1024 + k0 + c0];
      _Float16* l = &Bs[buf][r][c0];
      async_b128_off<0>(l, g);
      async_b128_off<16>(l, g);
      async_b128_off<32>(l, g);
      async_b128_off<48>(l, g);
    }
  };
  auto compute = [&](int buf) {
#pragma unroll
    for (int c = 0; c < 2; ++c) {
      half16 a = frag_a_rowmajor(&As[buf][mT * 16][c * 32], 64, l16, lh);
#pragma unroll
      for (int u = 0; u < 2; ++u) {
        half16 b = frag_b_nk(&Bs[buf][nT * 32 + u * 16][c * 32], 64, l16, lh);
        acc[u] = __builtin_amdgcn_wmma_f32_16x16x32_f16(
            false, a, false, b, (short)0, acc[u], false, false);
      }
    }
  };

  stage(0, 0);
  wait_async0();
  __syncthreads();
  for (int it = 0; it < 16; it += 2) {
    stage(1, (it + 1) * 64);
    compute(0);
    wait_async0();
    __syncthreads();
    if (it + 2 < 16) stage(0, (it + 2) * 64);
    compute(1);
    if (it + 2 < 16) {
      wait_async0();
      __syncthreads();
    }
  }

  // Epilogue: C layout lane l: col n=l&15, rows (l>>4)*8 + j
  const int mbase = bm * 32 + mT * 16 + lh * 8;
#pragma unroll
  for (int u = 0; u < 2; ++u) {
    const int nc = bn * 128 + nT * 32 + u * 16 + l16;  // [0,3072)
    const int seg = nc >> 10;                          // 0=Q 1=K 2=V
    const int cc = nc & 1023;
    const int h = cc >> 6, d = cc & 63;
    const float bias = bqkv[nc];
#pragma unroll
    for (int j = 0; j < 8; ++j) {
      int row = mbase + j;                             // [0,4096)
      int b = row >> 11, s = row & 2047;
      _Float16 hv = (_Float16)(acc[u][j] + bias);
      size_t head = (size_t)(b * 16 + h);
      if (seg == 0)       Qh[(head * 2048 + s) * 64 + d] = hv;
      else if (seg == 1)  Kh[(head * 2048 + s) * 64 + d] = hv;
      else                Vt[(head * 64 + d) * 2048 + s] = hv;
    }
  }
}

// ---------------------------------------------------------------------------
// Flash attention with multiplicative relative bias.
// Masked scores are 0 (not -inf) in the reference, so softmax covers ALL
// keys; every 32-key block is processed, masked elements zeroed.
// Block: 128 q-rows (8 waves x 16 rows) for one (b,h); K/V double-buffered.
// ---------------------------------------------------------------------------
__global__ __launch_bounds__(256) void k_attn(
    const _Float16* __restrict__ Qh, const _Float16* __restrict__ Kh,
    const _Float16* __restrict__ Vt, const float* __restrict__ rel_emb,
    const int* __restrict__ rel, _Float16* __restrict__ ah) {
  __shared__ _Float16 Ks[2][32][64];    // [key][d]  -> B for Q.K^T
  __shared__ _Float16 Vs[2][64][32];    // [d][key]  -> B for P.V
  __shared__ _Float16 Ps[8][16][32];    // per-wave P bounce (C -> A layout)

  const int bid = blockIdx.x;           // 512 = B * H * (S/128)
  const int qb = bid & 15;
  const int h = (bid >> 4) & 15;
  const int b = bid >> 8;
  const int tid = threadIdx.x;
  const int lane = tid & 31, wv = tid >> 5;
  const int l16 = lane & 15, lh = lane >> 4;
  const int qrow0 = qb * 128 + wv * 16;
  const size_t head = (size_t)(b * 16 + h);

  auto stageKV = [&](int buf, int kb) {
    {  // K tile 32x64
      int r = tid >> 3, c = (tid & 7) * 8;
      async_b128(&Ks[buf][r][c],
                 &Kh[(head * 2048 + kb * 32 + r) * 64 + c]);
    }
    {  // V^T tile 64x32
      int r = tid >> 2, c = (tid & 3) * 8;
      async_b128(&Vs[buf][r][c],
                 &Vt[(head * 64 + r) * 2048 + kb * 32 + c]);
    }
  };

  // Q fragments (16 rows x 64 d) loaded once, straight from global.
  half16 qa[2];
#pragma unroll
  for (int c = 0; c < 2; ++c)
    qa[c] = frag_a_rowmajor(&Qh[(head * 2048 + qrow0) * 64 + c * 32], 64,
                            l16, lh);

  floatx8 o[4] = {};                    // 16 rows x 64 d output accumulator
  float mrow[8], lrow[8];
#pragma unroll
  for (int j = 0; j < 8; ++j) { mrow[j] = -3.0e38f; lrow[j] = 0.0f; }

  auto body = [&](int buf, int kb) {
    // Scores: two 16x16 tiles (keys 0-15, 16-31), contract over d=64.
    floatx8 s0 = {}, s1 = {};
#pragma unroll
    for (int c = 0; c < 2; ++c) {
      half16 b0 = frag_b_nk(&Ks[buf][0][c * 32], 64, l16, lh);
      s0 = __builtin_amdgcn_wmma_f32_16x16x32_f16(false, qa[c], false, b0,
                                                  (short)0, s0, false, false);
      half16 b1 = frag_b_nk(&Ks[buf][16][c * 32], 64, l16, lh);
      s1 = __builtin_amdgcn_wmma_f32_16x16x32_f16(false, qa[c], false, b1,
                                                  (short)0, s1, false, false);
    }

    // Scale, multiplicative rel bias, mask (masked -> exactly 0).
#pragma unroll
    for (int j = 0; j < 8; ++j) {
      int qr = qrow0 + lh * 8 + j;
      long rbase = ((long)(b * 2048 + qr)) * 2048 + kb * 32;
      int k0c = kb * 32 + l16;
      int k1c = k0c + 16;
      int i0 = rel[rbase + l16];
      int i1 = rel[rbase + 16 + l16];
      float r0 = rel_emb[i0 * 16 + h];
      float r1 = rel_emb[i1 * 16 + h];
      s0[j] = (k0c <= qr) ? s0[j] * 0.125f * r0 : 0.0f;
      s1[j] = (k1c <= qr) ? s1[j] * 0.125f * r1 : 0.0f;
    }

    // Online softmax per row (rows live across 16-lane half-groups).
    float corr[8];
#pragma unroll
    for (int j = 0; j < 8; ++j) {
      float mx = fmaxf(s0[j], s1[j]);
      mx = fmaxf(mx, __shfl_xor(mx, 1, 16));
      mx = fmaxf(mx, __shfl_xor(mx, 2, 16));
      mx = fmaxf(mx, __shfl_xor(mx, 4, 16));
      mx = fmaxf(mx, __shfl_xor(mx, 8, 16));
      float mnew = fmaxf(mrow[j], mx);
      float c = __expf(mrow[j] - mnew);
      float p0 = __expf(s0[j] - mnew);
      float p1 = __expf(s1[j] - mnew);
      float rs = p0 + p1;
      rs += __shfl_xor(rs, 1, 16);
      rs += __shfl_xor(rs, 2, 16);
      rs += __shfl_xor(rs, 4, 16);
      rs += __shfl_xor(rs, 8, 16);
      lrow[j] = lrow[j] * c + rs;
      mrow[j] = mnew;
      corr[j] = c;
      s0[j] = p0;
      s1[j] = p1;
    }
#pragma unroll
    for (int t = 0; t < 4; ++t)
#pragma unroll
      for (int j = 0; j < 8; ++j) o[t][j] *= corr[j];

    // C-layout probs -> A-layout f16 via wave-private LDS bounce.
#pragma unroll
    for (int j = 0; j < 8; ++j) {
      int r = lh * 8 + j;
      Ps[wv][r][l16] = (_Float16)s0[j];
      Ps[wv][r][l16 + 16] = (_Float16)s1[j];
    }
    half16 pa = frag_a_rowmajor(&Ps[wv][0][0], 32, l16, lh);

    // P(16x32) . V(32x64): 4 WMMAs, one per 16-wide d tile.
#pragma unroll
    for (int t = 0; t < 4; ++t) {
      half16 vb = frag_b_nk(&Vs[buf][t * 16][0], 32, l16, lh);
      o[t] = __builtin_amdgcn_wmma_f32_16x16x32_f16(false, pa, false, vb,
                                                    (short)0, o[t], false,
                                                    false);
    }
  };

  stageKV(0, 0);
  wait_async0();
  __syncthreads();
  for (int kb = 0; kb < 64; kb += 2) {
    stageKV(1, kb + 1);
    body(0, kb);
    wait_async0();
    __syncthreads();
    if (kb + 2 < 64) stageKV(0, kb + 2);
    body(1, kb + 1);
    if (kb + 2 < 64) {
      wait_async0();
      __syncthreads();
    }
  }

  // Normalize and store merged-head activations (f16) for the proj GEMM.
#pragma unroll
  for (int t = 0; t < 4; ++t)
#pragma unroll
    for (int j = 0; j < 8; ++j) {
      int s = qrow0 + lh * 8 + j;
      float val = o[t][j] / lrow[j];
      ah[((size_t)(b * 2048 + s)) * 1024 + h * 64 + t * 16 + l16] =
          (_Float16)val;
    }
}

// ---------------------------------------------------------------------------
// Output projection: [4096 x 1024] @ [1024 x 1024] + bias -> f32 out.
// Same tiling as the QKV GEMM (M=32, N=128 per block).
// ---------------------------------------------------------------------------
__global__ __launch_bounds__(256) void k_proj_gemm(
    const _Float16* __restrict__ ah, const _Float16* __restrict__ WprojT,
    const float* __restrict__ bproj, float* __restrict__ out) {
  __shared__ _Float16 As[2][32][64];
  __shared__ _Float16 Bs[2][128][64];
  const int bm = blockIdx.x;  // 128
  const int bn = blockIdx.y;  // 8
  const int tid = threadIdx.x;
  const int lane = tid & 31, wv = tid >> 5;
  const int l16 = lane & 15, lh = lane >> 4;
  const int mT = wv & 1, nT = wv >> 1;

  floatx8 acc[2] = {};

  auto stage = [&](int buf, int k0) {
    {
      int r = tid >> 3, c = (tid & 7) * 8;
      async_b128(&As[buf][r][c],
                 &ah[((size_t)(bm * 32 + r)) * 1024 + k0 + c]);
    }
    {
      int r = tid >> 1, c0 = (tid & 1) * 32;
      const _Float16* g = &WprojT[((size_t)(bn * 128 + r)) * 1024 + k0 + c0];
      _Float16* l = &Bs[buf][r][c0];
      async_b128_off<0>(l, g);
      async_b128_off<16>(l, g);
      async_b128_off<32>(l, g);
      async_b128_off<48>(l, g);
    }
  };
  auto compute = [&](int buf) {
#pragma unroll
    for (int c = 0; c < 2; ++c) {
      half16 a = frag_a_rowmajor(&As[buf][mT * 16][c * 32], 64, l16, lh);
#pragma unroll
      for (int u = 0; u < 2; ++u) {
        half16 b = frag_b_nk(&Bs[buf][nT * 32 + u * 16][c * 32], 64, l16, lh);
        acc[u] = __builtin_amdgcn_wmma_f32_16x16x32_f16(
            false, a, false, b, (short)0, acc[u], false, false);
      }
    }
  };

  stage(0, 0);
  wait_async0();
  __syncthreads();
  for (int it = 0; it < 16; it += 2) {
    stage(1, (it + 1) * 64);
    compute(0);
    wait_async0();
    __syncthreads();
    if (it + 2 < 16) stage(0, (it + 2) * 64);
    compute(1);
    if (it + 2 < 16) {
      wait_async0();
      __syncthreads();
    }
  }

  const int mbase = bm * 32 + mT * 16 + lh * 8;
#pragma unroll
  for (int u = 0; u < 2; ++u) {
    const int nc = bn * 128 + nT * 32 + u * 16 + l16;
    const float bias = bproj[nc];
#pragma unroll
    for (int j = 0; j < 8; ++j)
      out[((size_t)(mbase + j)) * 1024 + nc] = acc[u][j] + bias;
  }
}

// ---------------------------------------------------------------------------
extern "C" void kernel_launch(void* const* d_in, const int* in_sizes, int n_in,
                              void* d_out, int out_size, void* d_ws,
                              size_t ws_size, hipStream_t stream) {
  const float* x       = (const float*)d_in[0];
  const float* Wqkv    = (const float*)d_in[1];
  const float* bqkv    = (const float*)d_in[2];
  const float* Wproj   = (const float*)d_in[3];
  const float* bproj   = (const float*)d_in[4];
  const float* rel_emb = (const float*)d_in[5];
  const int*   rel     = (const int*)d_in[6];
  float* out = (float*)d_out;

  // Workspace carve-out (~48 MB total).
  char* ws = (char*)d_ws;
  _Float16* xh     = (_Float16*)ws; ws += (size_t)4096 * 1024 * 2;
  _Float16* WqkvT  = (_Float16*)ws; ws += (size_t)3072 * 1024 * 2;
  _Float16* WprojT = (_Float16*)ws; ws += (size_t)1024 * 1024 * 2;
  _Float16* Qh     = (_Float16*)ws; ws += (size_t)4096 * 1024 * 2;
  _Float16* Kh     = (_Float16*)ws; ws += (size_t)4096 * 1024 * 2;
  _Float16* Vt     = (_Float16*)ws; ws += (size_t)4096 * 1024 * 2;
  _Float16* ahbuf  = (_Float16*)ws; ws += (size_t)4096 * 1024 * 2;

  k_transpose_cvt<<<dim3(32, 96), 256, 0, stream>>>(Wqkv, WqkvT, 3072);
  k_transpose_cvt<<<dim3(32, 32), 256, 0, stream>>>(Wproj, WprojT, 1024);
  k_cvt_f16<<<(4096 * 1024 / 4) / 256, 256, 0, stream>>>(x, xh,
                                                         4096 * 1024 / 4);
  k_qkv_gemm<<<dim3(128, 24), 256, 0, stream>>>(xh, WqkvT, bqkv, Qh, Kh, Vt);
  k_attn<<<512, 256, 0, stream>>>(Qh, Kh, Vt, rel_emb, rel, ahbuf);
  k_proj_gemm<<<dim3(128, 8), 256, 0, stream>>>(ahbuf, WprojT, bproj, out);
}
